// MetricLearningLoss_63247688401279
// MI455X (gfx1250) — matrix operand order repaired and checked by reading, not compiled, not run on hardware
//
#include <hip/hip_runtime.h>
#include <hip/hip_bf16.h>

typedef __attribute__((ext_vector_type(16))) __bf16 v16bf;
typedef __attribute__((ext_vector_type(8)))  float  v8f;

#define NROWS 4096
#define KDIM  1024

// loss constants: k = 4096, coeff = k/2 - 1, a = d2/(2k)/sigma^2, b = d2/(2k)/omega^2
#define COEFF      2047.0f
#define INV2K      (1.0f / 8192.0f)
#define INV_SIGMA2 25.0f            /* 1/0.2^2 */
#define EPSV       1e-12f

#define LDS_STRIDE 40               /* ushorts per row slice: 32 data + 8 pad = 80B = 20 DW (bank-conflict-free) */
#define LDS_BUF    (128 * LDS_STRIDE)

union FragU { uint4 q[2]; v16bf v; };

// pointer types matching the async-LDS builtin signature (from the round-2 diagnostic):
// param0: int __vector(4) in AS1 (global), param1: same pointee in AS3 (LDS)
typedef int v4i_vs __attribute__((vector_size(16)));
typedef __attribute__((address_space(1))) v4i_vs* as1_v4i_p;
typedef __attribute__((address_space(3))) v4i_vs* as3_v4i_p;

// ---- CDNA5 async global->LDS copy (16B per lane), ASYNCcnt-tracked ----
__device__ __forceinline__ void async_copy_b128(const unsigned short* gp, unsigned short* lp) {
#if __has_builtin(__builtin_amdgcn_global_load_async_to_lds_b128)
  __builtin_amdgcn_global_load_async_to_lds_b128(
      (as1_v4i_p)(gp), (as3_v4i_p)(lp), 0, 0);
#else
  asm volatile("global_load_async_to_lds_b128 %0, %1, off"
               :: "v"((unsigned)(size_t)lp), "v"((unsigned long long)(size_t)gp)
               : "memory");
#endif
}

__device__ __forceinline__ void wait_async0() {
#if __has_builtin(__builtin_amdgcn_s_wait_asynccnt)
  __builtin_amdgcn_s_wait_asynccnt(0);
#else
  asm volatile("s_wait_asynccnt 0x0" ::: "memory");
#endif
}

// A-matrix 16x32 bf16 layout (ISA 7.12.2): lane<16 -> row M=lane,    K = {0..7, 16..23}
//                                          lane>=16 -> row M=lane-16, K = {8..15, 24..31}
__device__ __forceinline__ v16bf lds_a_frag(const unsigned short* buf, int t16, int lane) {
  int half = lane >> 4;
  int r    = lane & 15;
  const unsigned short* p = buf + (16 * t16 + r) * LDS_STRIDE + half * 8;
  FragU f;
  f.q[0] = *(const uint4*)(p);
  f.q[1] = *(const uint4*)(p + 16);
  return f.v;
}

// B-matrix 32x16 bf16 layout: lane<16 -> col N=lane, K={0..7, 16..23} ; lane>=16 -> col N=lane-16, K={8..15, 24..31}
__device__ __forceinline__ v16bf lds_b_frag(const unsigned short* buf, int t16, int lane) {
  int half = lane >> 4;
  int c    = lane & 15;
  const unsigned short* p = buf + (16 * t16 + c) * LDS_STRIDE + half * 8;
  FragU f;
  f.q[0] = *(const uint4*)(p);
  f.q[1] = *(const uint4*)(p + 16);
  return f.v;
}

// ---------------- kernel 1: row sums of squares + fp32 -> (bf16 hi, bf16 lo) split ----------------
__global__ __launch_bounds__(256) void split_rows(const float* __restrict__ X,
                                                  unsigned short* __restrict__ hi,
                                                  unsigned short* __restrict__ lo,
                                                  float* __restrict__ sq) {
  __shared__ float red[256];
  const int row = blockIdx.x;
  const int t   = threadIdx.x;
  const float* xr = X + (size_t)row * KDIM;
  float acc = 0.0f;
  for (int c = t; c < KDIM; c += 256) {
    float x = xr[c];
    unsigned u  = __float_as_uint(x);
    unsigned uh = (u + 0x7FFFu + ((u >> 16) & 1u)) >> 16;   // RNE bf16
    float hf = __uint_as_float(uh << 16);
    float r  = x - hf;
    unsigned v  = __float_as_uint(r);
    unsigned ul = (v + 0x7FFFu + ((v >> 16) & 1u)) >> 16;
    hi[(size_t)row * KDIM + c] = (unsigned short)uh;
    lo[(size_t)row * KDIM + c] = (unsigned short)ul;
    acc += x * x;
  }
  red[t] = acc;
  __syncthreads();
  for (int s = 128; s > 0; s >>= 1) {
    if (t < s) red[t] += red[t + s];
    __syncthreads();
  }
  if (t == 0) sq[row] = red[0];
}

// ---------------- kernel 2: LDS-staged bf16x2 WMMA Gram tile + fused loss epilogue ----------------
__global__ __launch_bounds__(256) void gram_loss(const unsigned short* __restrict__ Xhi,
                                                 const unsigned short* __restrict__ Xlo,
                                                 const float* __restrict__ sq,
                                                 const int* __restrict__ labels,
                                                 float* __restrict__ partials) {
  const int tr  = blockIdx.y;
  const int tc  = blockIdx.x;
  const int bid = tr * gridDim.x + tc;
  const int t   = threadIdx.x;

  __shared__ unsigned short smem[4][LDS_BUF];   // 0:A-hi 1:A-lo 2:B-hi 3:B-lo  (40 KB)
  __shared__ float red[256];

  if (tc < tr) {                       // strictly-lower tile: contributes nothing
    if (t == 0) partials[bid] = 0.0f;  // ws is poisoned -> must write 0
    return;
  }

  const int wid  = t >> 5;
  const int lane = t & 31;
  const int wr   = wid >> 1;           // 4 wave-rows x 2 wave-cols
  const int wc   = wid & 1;
  const int row0 = tr * 128 + wr * 32; // wave tile: 32 rows x 64 cols
  const int col0 = tc * 128 + wc * 64;

  // cooperative async fill of all 4 buffers for k-slice k0:
  // wave -> (buffer, row-half); lane -> (row, 16B chunk); 8 instructions per wave
  auto stage = [&](int k0) {
    const int b     = wid >> 1;                       // 0..3
    const int rbase = (wid & 1) * 64;
    const unsigned short* src = (b & 1) ? Xlo : Xhi;
    const int grow0 = (b < 2 ? tr : tc) * 128;
    unsigned short* dst = &smem[b][0];
    const int chunk = lane & 3;                       // 16B chunk within 64B row-slice
    const int rl    = lane >> 2;                      // 8 rows per instruction
#pragma unroll
    for (int i = 0; i < 8; ++i) {
      int r = rbase + i * 8 + rl;
      async_copy_b128(src + (size_t)(grow0 + r) * KDIM + k0 + chunk * 8,
                      dst + r * LDS_STRIDE + chunk * 8);
    }
  };

  v8f acc[2][4];
#pragma unroll
  for (int m = 0; m < 2; ++m)
#pragma unroll
    for (int n = 0; n < 4; ++n) acc[m][n] = {};

  stage(0);
  wait_async0();
  __syncthreads();

  for (int k0 = 0; k0 < KDIM; k0 += 32) {
    v16bf ah[2], al[2], bh[4], bl[4];
#pragma unroll
    for (int m = 0; m < 2; ++m) {
      ah[m] = lds_a_frag(&smem[0][0], wr * 2 + m, lane);
      al[m] = lds_a_frag(&smem[1][0], wr * 2 + m, lane);
    }
#pragma unroll
    for (int n = 0; n < 4; ++n) {
      bh[n] = lds_b_frag(&smem[2][0], wc * 4 + n, lane);
      bl[n] = lds_b_frag(&smem[3][0], wc * 4 + n, lane);
    }
    asm volatile("s_wait_dscnt 0x0" ::: "memory");    // frags landed in VGPRs
    __syncthreads();                                   // LDS free to overwrite
    const bool more = (k0 + 32) < KDIM;
    if (more) stage(k0 + 32);                          // async copy overlaps WMMAs

    // three passes over independent accumulators: hi*hi, hi*lo, lo*hi
#pragma unroll
    for (int m = 0; m < 2; ++m)
#pragma unroll
      for (int n = 0; n < 4; ++n)
        acc[m][n] = __builtin_amdgcn_wmma_f32_16x16x32_bf16(
            false, ah[m], false, bh[n], (short)0, acc[m][n], false, false);
#pragma unroll
    for (int m = 0; m < 2; ++m)
#pragma unroll
      for (int n = 0; n < 4; ++n)
        acc[m][n] = __builtin_amdgcn_wmma_f32_16x16x32_bf16(
            false, ah[m], false, bl[n], (short)0, acc[m][n], false, false);
#pragma unroll
    for (int m = 0; m < 2; ++m)
#pragma unroll
      for (int n = 0; n < 4; ++n)
        acc[m][n] = __builtin_amdgcn_wmma_f32_16x16x32_bf16(
            false, al[m], false, bh[n], (short)0, acc[m][n], false, false);

    if (more) {
      wait_async0();
      __syncthreads();                                 // next slice visible to all waves
    }
  }

  // Epilogue. C/D layout: lane<16 -> (M=r, N=lane);  lane>=16 -> (M=8+r, N=lane-16)
  const int half = lane >> 4;
  const int lc   = lane & 15;

  float sqi[2][8]; int li[2][8];
#pragma unroll
  for (int m = 0; m < 2; ++m)
#pragma unroll
    for (int r = 0; r < 8; ++r) {
      int i = row0 + 16 * m + half * 8 + r;
      sqi[m][r] = sq[i];
      li[m][r]  = labels[i];
    }
  float sqj[4]; int lj[4];
#pragma unroll
  for (int n = 0; n < 4; ++n) {
    int j = col0 + 16 * n + lc;
    sqj[n] = sq[j];
    lj[n]  = labels[j];
  }

  float mysum = 0.0f;
#pragma unroll
  for (int m = 0; m < 2; ++m)
#pragma unroll
    for (int n = 0; n < 4; ++n)
#pragma unroll
      for (int r = 0; r < 8; ++r) {
        int i = row0 + 16 * m + half * 8 + r;
        int j = col0 + 16 * n + lc;
        if (j > i) {
          float g  = acc[m][n][r];
          float d2 = fmaxf(sqi[m][r] + sqj[n] - 2.0f * g, EPSV);
          float b  = d2 * INV2K;
          float a  = b * INV_SIGMA2;
          float term = (li[m][r] == lj[n])
                           ? (-COEFF * __logf(a) + 0.5f * a)
                           : ( COEFF * __logf(b) - 0.5f * b);
          mysum += term;
        }
      }

  red[t] = mysum;
  __syncthreads();
  for (int s = 128; s > 0; s >>= 1) {
    if (t < s) red[t] += red[t + s];
    __syncthreads();
  }
  if (t == 0) partials[bid] = red[0];
}

// ---------------- kernel 3: deterministic final reduction -> scalar ----------------
__global__ __launch_bounds__(256) void final_reduce(const float* __restrict__ partials,
                                                    float* __restrict__ out, int n) {
  __shared__ float red[256];
  int t = threadIdx.x;
  float s = 0.0f;
  for (int i = t; i < n; i += 256) s += partials[i];
  red[t] = s;
  __syncthreads();
  for (int k = 128; k > 0; k >>= 1) {
    if (t < k) red[t] += red[t + k];
    __syncthreads();
  }
  if (t == 0) out[0] = red[0];
}

extern "C" void kernel_launch(void* const* d_in, const int* in_sizes, int n_in,
                              void* d_out, int out_size, void* d_ws, size_t ws_size,
                              hipStream_t stream) {
  const float* X      = (const float*)d_in[0];
  const int*   labels = (const int*)d_in[1];

  unsigned short* Xhi = (unsigned short*)d_ws;
  unsigned short* Xlo = Xhi + (size_t)NROWS * KDIM;
  float* sq       = (float*)(Xlo + (size_t)NROWS * KDIM);
  float* partials = sq + NROWS;

  split_rows<<<NROWS, 256, 0, stream>>>(X, Xhi, Xlo, sq);

  dim3 grid(NROWS / 128, NROWS / 128);   // 32 x 32 tile grid
  gram_loss<<<grid, 256, 0, stream>>>(Xhi, Xlo, sq, labels, partials);

  final_reduce<<<1, 256, 0, stream>>>(partials, (float*)d_out,
                                      (NROWS / 128) * (NROWS / 128));
}
